// SATT_2_19447611916413
// MI455X (gfx1250) — matrix operand order repaired and checked by reading, not compiled, not run on hardware
//
#include <hip/hip_runtime.h>

#define BB 8
#define CC 128
#define NN 512
#define TT 12
#define HH 4
#define C4 32
#define BT (BB * TT)

typedef __attribute__((ext_vector_type(16))) __bf16 v16bf;
typedef __attribute__((ext_vector_type(8))) float v8f;
typedef __attribute__((ext_vector_type(4))) unsigned int u32x4;
typedef __attribute__((ext_vector_type(8))) int i32x8;
typedef __attribute__((ext_vector_type(4))) int i32x4;

union Frag {
  uint4 u[2];
  v16bf v;
};

static __device__ __forceinline__ unsigned short f2bf(float f) {
  unsigned int u = __float_as_uint(f);
  unsigned int r = (u + 0x7FFFu + ((u >> 16) & 1u)) >> 16;
  return (unsigned short)r;
}

#if defined(__gfx1250__) && __has_builtin(__builtin_amdgcn_tensor_load_to_lds) && \
    __has_builtin(__builtin_amdgcn_s_wait_tensorcnt)
#define USE_TDM 1
#else
#define USE_TDM 0
#endif

#if USE_TDM
// 2D TDM load: `nrows` rows (global stride `row_stride` bf16 elems) of
// `row_elems` contiguous bf16 each, into LDS with +16B pad every 64B
// (i.e. 32-bf16 rows padded to 40 shorts — matches the compute-side layout).
static __device__ __forceinline__ void tdm_load_2d(const unsigned short* gptr,
                                                   unsigned lds_off,
                                                   unsigned row_elems,
                                                   unsigned nrows,
                                                   unsigned row_stride) {
  unsigned long long ga = (unsigned long long)(const void*)gptr;
  u32x4 g0;
  g0[0] = 1u;                                  // count=1, is_restore=0, no gather
  g0[1] = lds_off;                             // lds_addr [63:32]
  g0[2] = (unsigned)(ga & 0xFFFFFFFFu);        // global_addr lo
  g0[3] = (unsigned)((ga >> 32) & 0x01FFFFFFu) // global_addr hi [120:96]
          | 0x80000000u;                       // type=2 in [127:126]
  i32x8 g1;
  // [17:16] data_size=1 (2B), [20] pad_enable, [24:22] pad_interval=3 (64B),
  // [31:25] pad_amount=3 (4 dwords = 16B)
  g1[0] = (int)((1u << 16) | (1u << 20) | (3u << 22) | (3u << 25));
  g1[1] = (int)((row_stride & 0xFFFFu) << 16);        // tensor_dim0 [79:48] lo
  g1[2] = (int)((row_stride >> 16) | (nrows << 16));  // dim0 hi | tensor_dim1 lo
  g1[3] = (int)(row_elems << 16);                     // tile_dim0 [127:112]
  g1[4] = (int)nrows;                                 // tile_dim1 [143:128]
  g1[5] = (int)row_stride;                            // tensor_dim0_stride lo
  g1[6] = 0;                                          // stride hi | dim1_stride lo
  g1[7] = 0;
  i32x4 z4 = {0, 0, 0, 0};
  i32x8 z8 = {0, 0, 0, 0, 0, 0, 0, 0};
  __builtin_amdgcn_tensor_load_to_lds(g0, g1, z4, z4, z8, 0);
}
#endif

// ---------------------------------------------------------------------------
// Stage 1: fused 1x1 convs.  f[o,n] = sum_c W[o,c] * seq[b,c,n,t]
// seq slice is staged ONCE, reused for both W1 and W2 (halves HBM reads).
// Output layout (bf16): F[bt][h][n][i]  with o = i*4 + h
// grid.x = 96 (b,t) * 8 (n-blocks of 64)
// ---------------------------------------------------------------------------
__global__ __launch_bounds__(256) void satt_stage1(
    const float* __restrict__ seq, const float* __restrict__ W1,
    const float* __restrict__ W2, unsigned short* __restrict__ F1,
    unsigned short* __restrict__ F2) {
  int wg = blockIdx.x;
  int nb = wg & 7;
  int bt = wg >> 3;  // 0..95
  int b = bt / TT, t = bt % TT;
  int n0 = nb * 64;

  __shared__ __align__(16) unsigned short wlds[128 * 136];  // [o][c], padded
  __shared__ __align__(16) unsigned short sq[64 * 136];     // [n_local][c], padded

  int tid = threadIdx.x;
  for (int idx = tid; idx < 64 * 128; idx += 256) {
    int n = idx & 63, c = idx >> 6;  // consecutive tid -> consecutive n
    sq[n * 136 + c] = f2bf(seq[((size_t)(b * CC + c) * NN + (n0 + n)) * TT + t]);
  }

  int lane = tid & 31, wave = tid >> 5;
  int row = lane & 15;
  int hi = (lane >= 16) ? 1 : 0;
  int ot = wave;  // 8 o-tiles, one per wave

  for (int which = 0; which < 2; ++which) {
    const float* W = which ? W2 : W1;
    unsigned short* F = which ? F2 : F1;
    __syncthreads();  // iter0: pair with load; iter1: protect wlds overwrite
    for (int idx = tid; idx < 128 * 128; idx += 256) {
      int o = idx >> 7, c = idx & 127;
      wlds[o * 136 + c] = f2bf(W[idx]);
    }
    __syncthreads();

    for (int nt = 0; nt < 4; ++nt) {
      v8f acc = {0.f, 0.f, 0.f, 0.f, 0.f, 0.f, 0.f, 0.f};
      for (int ks = 0; ks < 4; ++ks) {
        int kb = ks * 32;
        Frag a, bf;
        const unsigned short* ap = &wlds[(ot * 16 + row) * 136 + kb + hi * 8];
        a.u[0] = *(const uint4*)ap;
        a.u[1] = *(const uint4*)(ap + 16);
        const unsigned short* bp = &sq[(nt * 16 + row) * 136 + kb + hi * 8];
        bf.u[0] = *(const uint4*)bp;
        bf.u[1] = *(const uint4*)(bp + 16);
        acc = __builtin_amdgcn_wmma_f32_16x16x32_bf16(false, a.v, false, bf.v,
                                                      (short)0, acc, false, false);
      }
      int n = n0 + nt * 16 + row;
      for (int r = 0; r < 8; ++r) {
        int o = ot * 16 + r + hi * 8;
        int h = o & 3, i = o >> 2;
        F[((size_t)(bt * HH + h) * NN + n) * C4 + i] = f2bf(acc[r]);
      }
    }
  }
}

// ---------------------------------------------------------------------------
// Stage 2: per (b,t,h): logits = F1^T F2 (K = c4 = 32, one WMMA per tile),
// out[b,t,n,q] = mean_h sigmoid(logits).  LDS staged by the Tensor Data Mover
// (with hardware row padding), fallback = plain vector copy.
// grid.x = 96 (b,t) * 8 (n-blocks of 64) * 4 (q-blocks of 128)
// ---------------------------------------------------------------------------
__global__ __launch_bounds__(256) void satt_stage2(
    const unsigned short* __restrict__ F1, const unsigned short* __restrict__ F2,
    float* __restrict__ out) {
  int wg = blockIdx.x;
  int qb = wg & 3;
  int nbk = (wg >> 2) & 7;
  int bt = wg >> 5;  // 0..95
  int n0 = nbk * 64, q0 = qb * 128;

  __shared__ __align__(16) unsigned short lf1[HH * 64 * 40];   // 20 KB
  __shared__ __align__(16) unsigned short lf2[HH * 128 * 40];  // 40 KB

  int tid = threadIdx.x;
#if USE_TDM
  if (tid == 0) {
    // lf1: 4 h-chunks of 64*32 bf16, global h-stride 512*32 elems
    tdm_load_2d(&F1[((size_t)bt * HH * NN + n0) * C4],
                (unsigned)(unsigned long long)(void*)lf1, 64 * C4, HH, NN * C4);
    // lf2: 4 h-chunks of 128*32 bf16
    tdm_load_2d(&F2[((size_t)bt * HH * NN + q0) * C4],
                (unsigned)(unsigned long long)(void*)lf2, 128 * C4, HH, NN * C4);
  }
  if (tid < 32) __builtin_amdgcn_s_wait_tensorcnt((short)0);
  __syncthreads();
#else
  for (int idx = tid; idx < 256 * 4; idx += 256) {
    int rr = idx >> 2, ch = idx & 3;
    int h = rr >> 6, n = rr & 63;
    *(uint4*)&lf1[rr * 40 + ch * 8] =
        *(const uint4*)&F1[((size_t)(bt * HH + h) * NN + n0 + n) * C4 + ch * 8];
  }
  for (int idx = tid; idx < 512 * 4; idx += 256) {
    int rr = idx >> 2, ch = idx & 3;
    int h = rr >> 7, q = rr & 127;
    *(uint4*)&lf2[rr * 40 + ch * 8] =
        *(const uint4*)&F2[((size_t)(bt * HH + h) * NN + q0 + q) * C4 + ch * 8];
  }
  __syncthreads();
#endif

  int lane = tid & 31, wave = tid >> 5;
  int row = lane & 15;
  int hi = (lane >= 16) ? 1 : 0;
  int mt = wave & 3;          // row tile within 64-row block
  int qt0 = (wave >> 2) * 4;  // each wave covers 4 q-tiles

  Frag a[HH];
  for (int h = 0; h < HH; ++h) {
    const unsigned short* ap = &lf1[(h * 64 + mt * 16 + row) * 40 + hi * 8];
    a[h].u[0] = *(const uint4*)ap;
    a[h].u[1] = *(const uint4*)(ap + 16);
  }

  for (int j = 0; j < 4; ++j) {
    int qt = qt0 + j;
    float sum[8] = {0.f, 0.f, 0.f, 0.f, 0.f, 0.f, 0.f, 0.f};
    for (int h = 0; h < HH; ++h) {
      Frag bf;
      const unsigned short* bp = &lf2[(h * 128 + qt * 16 + row) * 40 + hi * 8];
      bf.u[0] = *(const uint4*)bp;
      bf.u[1] = *(const uint4*)(bp + 16);
      v8f z = {0.f, 0.f, 0.f, 0.f, 0.f, 0.f, 0.f, 0.f};
      v8f d = __builtin_amdgcn_wmma_f32_16x16x32_bf16(false, a[h].v, false, bf.v,
                                                      (short)0, z, false, false);
      for (int r = 0; r < 8; ++r) sum[r] += 1.0f / (1.0f + __expf(-d[r]));
    }
    int q = q0 + qt * 16 + row;
    for (int r = 0; r < 8; ++r) {
      int n = n0 + mt * 16 + r + hi * 8;
      out[((size_t)bt * NN + n) * NN + q] = 0.25f * sum[r];
    }
  }
}

extern "C" void kernel_launch(void* const* d_in, const int* in_sizes, int n_in,
                              void* d_out, int out_size, void* d_ws, size_t ws_size,
                              hipStream_t stream) {
  const float* seq = (const float*)d_in[0];
  const float* W1 = (const float*)d_in[1];
  const float* W2 = (const float*)d_in[2];
  float* out = (float*)d_out;
  unsigned short* F1 = (unsigned short*)d_ws;
  unsigned short* F2 = F1 + (size_t)BT * HH * NN * C4;

  satt_stage1<<<dim3(768), 256, 0, stream>>>(seq, W1, W2, F1, F2);
  satt_stage2<<<dim3(3072), 256, 0, stream>>>(F1, F2, out);
}
